// SmolAttention_40346922779456
// MI455X (gfx1250) — compile-verified, matrix operand order checked by reading, not run on hardware
//
#include <hip/hip_runtime.h>
#include <hip/hip_bf16.h>

// ---------------------------------------------------------------------------
// Problem constants (from reference): B=4, S=2048, H=576, NH=9, NKV=3, HD=64
// ---------------------------------------------------------------------------
constexpr int Bc   = 4;
constexpr int Sc   = 2048;
constexpr int Hc   = 576;
constexpr int NHc  = 9;
constexpr int NKVc = 3;
constexpr int HDc  = 64;
constexpr int GROUPSc = NHc / NKVc;            // 3
constexpr int Mrows = Bc * Sc;                 // 8192
constexpr int Nqkv  = (NHc + 2 * NKVc) * HDc;  // 960 (Q:576 | K:192 | V:192)
constexpr float ROPE_THETA = 100000.0f;

typedef __attribute__((ext_vector_type(16))) _Float16 v16h;
typedef __attribute__((ext_vector_type(8)))  _Float16 v8h;
typedef __attribute__((ext_vector_type(8)))  float    v8f;

// Workspace layout (bytes). Total ~66 MB.
constexpr size_t OFF_HS    = 0;                                  // f16 [8192][576]
constexpr size_t OFF_WQKV  = OFF_HS    + (size_t)Mrows*Hc*2;     // f16 [960][576]
constexpr size_t OFF_WO    = OFF_WQKV  + (size_t)Nqkv*Hc*2;      // f16 [576][576]
constexpr size_t OFF_COS   = OFF_WO    + (size_t)Hc*Hc*2;        // f32 [2048][64]
constexpr size_t OFF_SIN   = OFF_COS   + (size_t)Sc*HDc*4;
constexpr size_t OFF_QKV32 = OFF_SIN   + (size_t)Sc*HDc*4;       // f32 [8192][960]
constexpr size_t OFF_QF    = OFF_QKV32 + (size_t)Mrows*Nqkv*4;   // f16 [B][9][S][64]
constexpr size_t OFF_KF    = OFF_QF    + (size_t)Bc*NHc*Sc*HDc*2;// f16 [B][3][S][64]
constexpr size_t OFF_VF    = OFF_KF    + (size_t)Bc*NKVc*Sc*HDc*2;
constexpr size_t OFF_ATTN  = OFF_VF    + (size_t)Bc*NKVc*Sc*HDc*2;// f16 [8192][576]

// ---------------------------------------------------------------------------
// Fragment loader for the CDNA5 16-bit A-layout (16x32, wave32):
//   lane<16 : row = lane,    K = {k0+0..7} in v[0..7],  {k0+16..23} in v[8..15]
//   lane>=16: row = lane-16, K = {k0+8..15},            {k0+24..31}
// Two 16-byte loads per lane.  The B operand of D = A x B^T (both row-major)
// uses the exact same pattern on the other matrix's rows.
// ---------------------------------------------------------------------------
template <typename PT>
static __device__ __forceinline__ v16h load_frag_a(PT base, int ld, int k0) {
    int lane = threadIdx.x & 31;
    int row  = lane & 15;
    int kb   = k0 + ((lane >> 4) << 3);
    PT p = base + (size_t)row * ld + kb;
    v8h lo = *(const v8h*)(p);
    v8h hi = *(const v8h*)(p + 16);
    v16h r;
#pragma unroll
    for (int i = 0; i < 8; ++i) { r[i] = lo[i]; r[i + 8] = hi[i]; }
    return r;
}

static __device__ __forceinline__ v8f wmma_f16(v16h a, v16h b, v8f c) {
    return __builtin_amdgcn_wmma_f32_16x16x32_f16(false, a, false, b, (short)0, c,
                                                  false, false);
}

// ---------------------------------------------------------------------------
// Kernel 1: f32 -> f16 conversions + RoPE tables.
// ---------------------------------------------------------------------------
__global__ void prep_kernel(const float* __restrict__ hs,
                            const float* __restrict__ Wq,
                            const float* __restrict__ Wk,
                            const float* __restrict__ Wv,
                            const float* __restrict__ Wo,
                            _Float16* __restrict__ hsF16,
                            _Float16* __restrict__ wqkvF16,
                            _Float16* __restrict__ woF16,
                            float* __restrict__ cosT,
                            float* __restrict__ sinT) {
    const size_t nHS = (size_t)Mrows * Hc;
    const size_t nWq = (size_t)NHc  * HDc * Hc;
    const size_t nWk = (size_t)NKVc * HDc * Hc;
    const size_t nWv = nWk;
    const size_t nWo = (size_t)Hc * Hc;
    const size_t nRope = (size_t)Sc * HDc;
    const size_t total = nHS + nWq + nWk + nWv + nWo + nRope;
    size_t stride = (size_t)gridDim.x * blockDim.x;
    for (size_t i = (size_t)blockIdx.x * blockDim.x + threadIdx.x; i < total; i += stride) {
        size_t j = i;
        if (j < nHS) { hsF16[j] = (_Float16)hs[j]; continue; }
        j -= nHS;
        if (j < nWq) { wqkvF16[j] = (_Float16)Wq[j]; continue; }
        j -= nWq;
        if (j < nWk) { wqkvF16[nWq + j] = (_Float16)Wk[j]; continue; }
        j -= nWk;
        if (j < nWv) { wqkvF16[nWq + nWk + j] = (_Float16)Wv[j]; continue; }
        j -= nWv;
        if (j < nWo) { woF16[j] = (_Float16)Wo[j]; continue; }
        j -= nWo;
        // RoPE table: position s, dim d; angle = s * theta^(-2*(d%32)/64)
        int s = (int)(j >> 6);
        int d = (int)(j & 63);
        float inv = __expf(-(2.0f * (float)(d & 31) / (float)HDc) * __logf(ROPE_THETA));
        float ang = (float)s * inv;
        cosT[j] = __cosf(ang);
        sinT[j] = __sinf(ang);
    }
}

// ---------------------------------------------------------------------------
// Kernel 2/5: WMMA GEMM  C[M][N] = A[M][K] * Bw[N][K]^T   (f16 in, f32 out)
// Block = 128 threads (4 waves stacked along M); block tile 128x64, wave tile
// 32x64 (2x4 accum frags).  B fragments are identical across the 4 waves ->
// L0 reuse.  grid.x = M/128, grid.y = N/64.  All dims exact multiples.
// ---------------------------------------------------------------------------
__global__ __launch_bounds__(128) void gemm_f16_kernel(
    const _Float16* __restrict__ A, const _Float16* __restrict__ Bw,
    float* __restrict__ C, int N, int K) {
    const int wave = threadIdx.x >> 5;
    const int mbase = blockIdx.x * 128 + wave * 32;
    const int nbase = blockIdx.y * 64;
    const _Float16* Ar0 = A + (size_t)mbase * K;
    const _Float16* Ar1 = A + (size_t)(mbase + 16) * K;
    const _Float16* Br[4] = {
        Bw + (size_t)(nbase +  0) * K, Bw + (size_t)(nbase + 16) * K,
        Bw + (size_t)(nbase + 32) * K, Bw + (size_t)(nbase + 48) * K};
    v8f acc[2][4] = {};
    for (int k0 = 0; k0 < K; k0 += 32) {
        if (k0 + 32 < K) {  // warm caches for the next K-slab
            __builtin_prefetch(Ar0 + k0 + 32, 0, 1);
            __builtin_prefetch(Ar1 + k0 + 32, 0, 1);
        }
        v16h a0 = load_frag_a(Ar0, K, k0);
        v16h a1 = load_frag_a(Ar1, K, k0);
#pragma unroll
        for (int j = 0; j < 4; ++j) {
            v16h b = load_frag_a(Br[j], K, k0);
            acc[0][j] = wmma_f16(a0, b, acc[0][j]);
            acc[1][j] = wmma_f16(a1, b, acc[1][j]);
        }
    }
    // C frag layout: VGPR r, lane l -> row r + 8*(l>>4), col l&15
    const int lane = threadIdx.x & 31;
    const int col = lane & 15;
    const int roff = (lane >> 4) * 8;
#pragma unroll
    for (int i = 0; i < 2; ++i) {
#pragma unroll
        for (int j = 0; j < 4; ++j) {
#pragma unroll
            for (int r = 0; r < 8; ++r) {
                C[(size_t)(mbase + i * 16 + roff + r) * N +
                  (nbase + j * 16 + col)] = acc[i][j][r];
            }
        }
    }
}

// ---------------------------------------------------------------------------
// Kernel 3: RoPE + relayout to head-major f16 Q/K/V.
// grid = B*S blocks of 64 threads (thread = head dim d).
// ---------------------------------------------------------------------------
__global__ __launch_bounds__(64) void rope_kernel(
    const float* __restrict__ qkv, const float* __restrict__ cosT,
    const float* __restrict__ sinT, _Float16* __restrict__ qF16,
    _Float16* __restrict__ kF16, _Float16* __restrict__ vF16) {
    const int bs = blockIdx.x;
    const int b = bs / Sc, s = bs % Sc;
    const int d = threadIdx.x;
    const float* row = qkv + (size_t)bs * Nqkv;
    const float c  = cosT[s * HDc + d];
    const float sn = sinT[s * HDc + d];
#pragma unroll
    for (int h = 0; h < NHc; ++h) {
        float x  = row[h * HDc + d];
        float xr = (d < 32) ? -row[h * HDc + d + 32] : row[h * HDc + d - 32];
        qF16[(((size_t)b * NHc + h) * Sc + s) * HDc + d] = (_Float16)(x * c + xr * sn);
    }
#pragma unroll
    for (int h = 0; h < NKVc; ++h) {
        float x  = row[NHc * HDc + h * HDc + d];
        float xr = (d < 32) ? -row[NHc * HDc + h * HDc + d + 32]
                            :  row[NHc * HDc + h * HDc + d - 32];
        size_t o = (((size_t)b * NKVc + h) * Sc + s) * HDc + d;
        kF16[o] = (_Float16)(x * c + xr * sn);
        vF16[o] = (_Float16)row[(NHc + NKVc) * HDc + h * HDc + d];
    }
}

// ---------------------------------------------------------------------------
// Kernel 4: causal flash attention.
// grid.x = S/64 (q tiles), grid.y = B*NH. Block = 128 threads (4 waves),
// wave owns 16 q rows; KV stepped by 32 keys.  V tile is staged into LDS with
// gfx1250 async-to-LDS copies (ASYNCcnt), transposed in LDS, then consumed as
// WMMA B fragments.  Online softmax in f32 with 16-lane butterfly reductions.
// ---------------------------------------------------------------------------
__global__ __launch_bounds__(128) void attn_kernel(
    const _Float16* __restrict__ qF16, const _Float16* __restrict__ kF16,
    const _Float16* __restrict__ vF16, _Float16* __restrict__ attnOut) {
    __shared__ _Float16 vstage[32 * HDc];    // raw V tile [key][dim]   (4 KB)
    __shared__ _Float16 vt[HDc * 32];        // V^T tile  [dim][key]    (4 KB)
    __shared__ _Float16 pbuf[4][16 * 32];    // per-wave P relayout     (4 KB)

    const int bh = blockIdx.y;
    const int b = bh / NHc, h = bh % NHc, kvh = h / GROUPSc;
    const _Float16* Q  = qF16 + (size_t)bh * Sc * HDc;
    const _Float16* Kh = kF16 + (size_t)(b * NKVc + kvh) * Sc * HDc;
    const _Float16* Vh = vF16 + (size_t)(b * NKVc + kvh) * Sc * HDc;

    const int wave = threadIdx.x >> 5;
    const int lane = threadIdx.x & 31;
    const int q0 = blockIdx.x * 64 + wave * 16;

    v16h qf0 = load_frag_a(Q + (size_t)q0 * HDc, HDc, 0);
    v16h qf1 = load_frag_a(Q + (size_t)q0 * HDc, HDc, 32);

    v8f o0 = {}, o1 = {}, o2 = {}, o3 = {};
    float rmax[8], rsum[8];
#pragma unroll
    for (int r = 0; r < 8; ++r) { rmax[r] = -1e30f; rsum[r] = 0.0f; }

    const int col  = lane & 15;
    const int roff = (lane >> 4) * 8;
    const int kv_max = blockIdx.x * 64 + 64;   // uniform across the block
    const int kk = threadIdx.x & 31;           // transpose: key within tile
    const int dg = threadIdx.x >> 5;           // transpose: 16-dim group
    // LDS byte offset of vstage for this thread's async destination: generic
    // LDS pointers carry the LDS byte offset in addr[31:0] (aperture encoding).
    const unsigned vstage_lds =
        (unsigned)(uintptr_t)(&vstage[0]) + (unsigned)threadIdx.x * 32u;

    for (int kv = 0; kv < kv_max; kv += 32) {
        // --- async copy of V tile (4 KB) into LDS: 128 thr x 32 B each ---
        {
            const _Float16* g = Vh + (size_t)kv * HDc + (size_t)threadIdx.x * 16;
            asm volatile(
                "global_load_async_to_lds_b128 %0, %1, off\n\t"
                "global_load_async_to_lds_b128 %0, %1, off offset:16"
                :
                : "v"(vstage_lds), "v"(g)
                : "memory");
        }
        asm volatile("s_wait_asynccnt 0" ::: "memory");
        __syncthreads();
        // --- in-LDS transpose vstage[key][dim] -> vt[dim][key] ---
        {
            const _Float16* src = &vstage[0] + kk * HDc + dg * 16;
            v8h x0 = *(const v8h*)(src);
            v8h x1 = *(const v8h*)(src + 8);
#pragma unroll
            for (int j = 0; j < 8; ++j) {
                vt[(dg * 16 + j) * 32 + kk]     = x0[j];
                vt[(dg * 16 + j + 8) * 32 + kk] = x1[j];
            }
        }
        __syncthreads();

        const bool active = (kv < q0 + 16);   // wave-uniform
        if (active) {
            // --- scores = Q(16x64) * K_tile^T : 4 WMMA ---
            v8f sc0 = {}, sc1 = {};
            sc0 = wmma_f16(qf0, load_frag_a(Kh + (size_t)kv * HDc, HDc, 0), sc0);
            sc0 = wmma_f16(qf1, load_frag_a(Kh + (size_t)kv * HDc, HDc, 32), sc0);
            sc1 = wmma_f16(qf0, load_frag_a(Kh + (size_t)(kv + 16) * HDc, HDc, 0), sc1);
            sc1 = wmma_f16(qf1, load_frag_a(Kh + (size_t)(kv + 16) * HDc, HDc, 32), sc1);

            // scale + causal mask (mask input == analytic causal mask)
#pragma unroll
            for (int r = 0; r < 8; ++r) {
                int qrow = q0 + roff + r;
                sc0[r] = (kv + col      > qrow) ? -1e30f : sc0[r] * 0.125f;
                sc1[r] = (kv + 16 + col > qrow) ? -1e30f : sc1[r] * 0.125f;
            }
            // --- online softmax ---
            float alpha[8];
#pragma unroll
            for (int r = 0; r < 8; ++r) {
                float m = fmaxf(sc0[r], sc1[r]);
#pragma unroll
                for (int off = 1; off < 16; off <<= 1)
                    m = fmaxf(m, __shfl_xor(m, off, 32));
                float mn = fmaxf(rmax[r], m);
                alpha[r] = __expf(rmax[r] - mn);
                rmax[r] = mn;
                float p0 = __expf(sc0[r] - mn);
                float p1 = __expf(sc1[r] - mn);
                sc0[r] = p0; sc1[r] = p1;
                float rs = p0 + p1;
#pragma unroll
                for (int off = 1; off < 16; off <<= 1)
                    rs += __shfl_xor(rs, off, 32);
                rsum[r] = rsum[r] * alpha[r] + rs;
                o0[r] *= alpha[r]; o1[r] *= alpha[r];
                o2[r] *= alpha[r]; o3[r] *= alpha[r];
            }
            // --- relayout P (C-layout) -> A-layout via per-wave LDS ---
            _Float16* pb = &pbuf[wave][0];
#pragma unroll
            for (int r = 0; r < 8; ++r) {
                pb[(roff + r) * 32 + col]      = (_Float16)sc0[r];
                pb[(roff + r) * 32 + 16 + col] = (_Float16)sc1[r];
            }
        }
        __syncthreads();   // orders P stores before P loads; vt stable

        if (active) {
            v16h pa = load_frag_a((const _Float16*)&pbuf[wave][0], 32, 0);
            o0 = wmma_f16(pa, load_frag_a((const _Float16*)vt +  0 * 32, 32, 0), o0);
            o1 = wmma_f16(pa, load_frag_a((const _Float16*)vt + 16 * 32, 32, 0), o1);
            o2 = wmma_f16(pa, load_frag_a((const _Float16*)vt + 32 * 32, 32, 0), o2);
            o3 = wmma_f16(pa, load_frag_a((const _Float16*)vt + 48 * 32, 32, 0), o3);
        }
        __syncthreads();   // protect vstage/vt before next tile's overwrite
    }

    // --- normalize and store to attn buffer [b][s][h*64 + d] (f16) ---
    _Float16* outp = attnOut + ((size_t)b * Sc) * (NHc * HDc) + h * HDc;
    v8f* of[4] = {&o0, &o1, &o2, &o3};
#pragma unroll
    for (int r = 0; r < 8; ++r) {
        float inv = 1.0f / rsum[r];
        int s = q0 + roff + r;
#pragma unroll
        for (int f = 0; f < 4; ++f) {
            outp[(size_t)s * (NHc * HDc) + f * 16 + col] = (_Float16)((*of[f])[r] * inv);
        }
    }
}

// ---------------------------------------------------------------------------
// Host-side launch.
// Inputs: 0=hidden_states(f32) 1=attention_mask(unused; analytic causal)
//         2=position_ids(unused; arange) 3=Wq 4=Wk 5=Wv 6=Wo
// ---------------------------------------------------------------------------
extern "C" void kernel_launch(void* const* d_in, const int* in_sizes, int n_in,
                              void* d_out, int out_size, void* d_ws, size_t ws_size,
                              hipStream_t stream) {
    (void)in_sizes; (void)n_in; (void)out_size; (void)ws_size;
    const float* hs = (const float*)d_in[0];
    const float* Wq = (const float*)d_in[3];
    const float* Wk = (const float*)d_in[4];
    const float* Wv = (const float*)d_in[5];
    const float* Wo = (const float*)d_in[6];
    float* out = (float*)d_out;

    char* ws = (char*)d_ws;
    _Float16* hsF16   = (_Float16*)(ws + OFF_HS);
    _Float16* wqkvF16 = (_Float16*)(ws + OFF_WQKV);
    _Float16* woF16   = (_Float16*)(ws + OFF_WO);
    float*    cosT    = (float*)(ws + OFF_COS);
    float*    sinT    = (float*)(ws + OFF_SIN);
    float*    qkv32   = (float*)(ws + OFF_QKV32);
    _Float16* qF16    = (_Float16*)(ws + OFF_QF);
    _Float16* kF16    = (_Float16*)(ws + OFF_KF);
    _Float16* vF16    = (_Float16*)(ws + OFF_VF);
    _Float16* attnF16 = (_Float16*)(ws + OFF_ATTN);

    prep_kernel<<<2048, 256, 0, stream>>>(hs, Wq, Wk, Wv, Wo, hsF16, wqkvF16,
                                          woF16, cosT, sinT);

    dim3 gQKV(Mrows / 128, Nqkv / 64);  // 64 x 15
    gemm_f16_kernel<<<gQKV, 128, 0, stream>>>(hsF16, wqkvF16, qkv32, Nqkv, Hc);

    rope_kernel<<<Mrows, 64, 0, stream>>>(qkv32, cosT, sinT, qF16, kF16, vF16);

    dim3 gA(Sc / 64, Bc * NHc);         // 32 x 36
    attn_kernel<<<gA, 128, 0, stream>>>(qF16, kF16, vF16, attnF16);

    dim3 gO(Mrows / 128, Hc / 64);      // 64 x 9
    gemm_f16_kernel<<<gO, 128, 0, stream>>>(attnF16, woF16, out, Hc, Hc);
}
